// GraphConvolutionalNetwork_31009663877670
// MI455X (gfx1250) — matrix operand order repaired and checked by reading, not compiled, not run on hardware
//
#include <hip/hip_runtime.h>
#include <hip/hip_bf16.h>

typedef __attribute__((ext_vector_type(8)))  float  v8f;
typedef __attribute__((ext_vector_type(16))) __bf16 v16bf;
typedef __attribute__((ext_vector_type(8)))  __bf16 v8bf;

#define N_NODES  40000
#define N_EDGES  640000
#define DIM      128
#define N_GRAPHS 128
#define LDSS     136   // padded LDS row stride in bf16 elems (272B, 16B aligned, conflict-free)

// ---------------------------------------------------------------- fill
__global__ void fill_kernel(float* __restrict__ p, float val, int n) {
    int i = blockIdx.x * blockDim.x + threadIdx.x;
    int stride = gridDim.x * blockDim.x;
    for (; i < n; i += stride) p[i] = val;
}

// ------------------------------------------- weight convert + transpose
// Wt[n*128 + k] = (bf16) W[k*128 + n]   (so B fragments are contiguous in K)
__global__ void convert_weight_kernel(const float* __restrict__ W,
                                      __bf16* __restrict__ Wt) {
    int id = blockIdx.x * blockDim.x + threadIdx.x;   // 0 .. 16383
    int n = id >> 7;
    int k = id & 127;
    Wt[id] = (__bf16)W[k * DIM + n];
}

// ---------------------------------------------------------- scatter add
// one wave (32 lanes) per edge; each lane owns 4 consecutive floats
__global__ void scatter_add_kernel(const float* __restrict__ x,
                                   const int* __restrict__ src,
                                   const int* __restrict__ dst,
                                   float* __restrict__ agg) {
    int gid  = blockIdx.x * blockDim.x + threadIdx.x;
    int e    = gid >> 5;
    int lane = gid & 31;
    if (e >= N_EDGES) return;
    int s = src[e];
    int d = dst[e];
    const float4 v = *reinterpret_cast<const float4*>(x + (size_t)s * DIM + lane * 4);
    float* a = agg + (size_t)d * DIM + lane * 4;
    atomicAdd(a + 0, v.x);
    atomicAdd(a + 1, v.y);
    atomicAdd(a + 2, v.z);
    atomicAdd(a + 3, v.w);
}

// ----------------------------------------------------- fused GraphConv
// out[16 x 128 stripe] = relu?( agg @ WtRel^T + b + x @ WtRoot^T )
// block = 256 threads = 8 waves; wave w owns N-tile w (16 cols); 4 K-steps
// of v_wmma_f32_16x16x32_bf16 per operand matrix, accumulated in one v8f.
__global__ void __launch_bounds__(256)
graphconv_gemm_kernel(const float* __restrict__ agg,
                      const float* __restrict__ xin,
                      const __bf16* __restrict__ WtRel,
                      const __bf16* __restrict__ WtRoot,
                      const float* __restrict__ bias,
                      float* __restrict__ out,
                      int do_relu) {
    __shared__ __align__(16) __bf16 sT[2][16 * LDSS]; // [0]=agg rows, [1]=x rows (bf16)

    const int row0 = blockIdx.x * 16;
    const int tid  = threadIdx.x;

    // stage 16x128 stripes of agg and x into LDS as bf16 (coalesced f32 loads)
    for (int i = tid; i < 16 * DIM; i += 256) {
        int m = i >> 7;
        int c = i & 127;
        sT[0][m * LDSS + c] = (__bf16)agg[(size_t)(row0 + m) * DIM + c];
        sT[1][m * LDSS + c] = (__bf16)xin[(size_t)(row0 + m) * DIM + c];
    }
    __syncthreads();

    const int lane = tid & 31;
    const int wave = tid >> 5;       // N-tile index 0..7
    const int nl   = lane & 15;      // A: row m ; B/C/D: col n (within tile)
    const int half = lane >> 4;
    const int n_g  = wave * 16 + nl; // global output column

    v8f acc = {};

    for (int pass = 0; pass < 2; ++pass) {
        const __bf16* Wt = pass ? WtRoot : WtRel;
        for (int kk = 0; kk < 4; ++kk) {
            // A fragment (16x32 bf16): VGPR0-3 = K[ab..ab+7], VGPR4-7 = K[ab+16..ab+23]
            int ab = kk * 32 + half * 8;
            v8bf a0 = *reinterpret_cast<const v8bf*>(&sT[pass][nl * LDSS + ab]);
            v8bf a1 = *reinterpret_cast<const v8bf*>(&sT[pass][nl * LDSS + ab + 16]);
            // B fragment (32x16 bf16): VGPR0-3 = K[bb..bb+7], VGPR4-7 = K[bb+16..bb+23]
            int bb = kk * 32 + half * 8;
            v8bf b0 = *reinterpret_cast<const v8bf*>(Wt + (size_t)n_g * DIM + bb);
            v8bf b1 = *reinterpret_cast<const v8bf*>(Wt + (size_t)n_g * DIM + bb + 16);
            v16bf a = __builtin_shufflevector(a0, a1, 0,1,2,3,4,5,6,7,8,9,10,11,12,13,14,15);
            v16bf b = __builtin_shufflevector(b0, b1, 0,1,2,3,4,5,6,7,8,9,10,11,12,13,14,15);
            acc = __builtin_amdgcn_wmma_f32_16x16x32_bf16(
                      /*neg_a=*/false, a, /*neg_b=*/false, b,
                      /*c_mod=*/(short)0, acc, /*reuse_a=*/false, /*reuse_b=*/false);
        }
    }

    // epilogue: bias + optional relu; C/D layout: VGPR r -> row (r + half*8), col n_g
    float bv = bias[n_g];
    int mbase = row0 + half * 8;
    for (int r = 0; r < 8; ++r) {
        float v = acc[r] + bv;
        if (do_relu) v = fmaxf(v, 0.0f);
        out[(size_t)(mbase + r) * DIM + n_g] = v;
    }
}

// ------------------------------------------------------------- pooling
__device__ __forceinline__ void atomic_max_f32(float* addr, float v) {
    int iv = __float_as_int(v);
    if (iv >= 0)
        atomicMax(reinterpret_cast<int*>(addr), iv);
    else
        atomicMin(reinterpret_cast<unsigned int*>(addr), __float_as_uint(v));
}

__global__ void pool_max_kernel(const float* __restrict__ h,
                                const int* __restrict__ batch,
                                float* __restrict__ out) {
    int gid  = blockIdx.x * blockDim.x + threadIdx.x;
    int node = gid >> 5;
    int lane = gid & 31;
    if (node >= N_NODES) return;
    int g = batch[node];
    const float4 v = *reinterpret_cast<const float4*>(h + (size_t)node * DIM + lane * 4);
    float* o = out + (size_t)g * DIM + lane * 4;
    atomic_max_f32(o + 0, v.x);
    atomic_max_f32(o + 1, v.y);
    atomic_max_f32(o + 2, v.z);
    atomic_max_f32(o + 3, v.w);
}

// -------------------------------------------------------------- launch
extern "C" void kernel_launch(void* const* d_in, const int* in_sizes, int n_in,
                              void* d_out, int out_size, void* d_ws, size_t ws_size,
                              hipStream_t stream) {
    const float* x     = (const float*)d_in[0];
    const int*   edge  = (const int*)d_in[1];
    const int*   batch = (const int*)d_in[2];
    const float* W_rel[3]  = {(const float*)d_in[3], (const float*)d_in[6], (const float*)d_in[9]};
    const float* b_rel[3]  = {(const float*)d_in[4], (const float*)d_in[7], (const float*)d_in[10]};
    const float* W_root[3] = {(const float*)d_in[5], (const float*)d_in[8], (const float*)d_in[11]};

    const size_t NF = (size_t)N_NODES * DIM;
    char* ws = (char*)d_ws;
    float* agg = (float*)ws;  ws += NF * sizeof(float);
    float* h1  = (float*)ws;  ws += NF * sizeof(float);
    float* h2  = (float*)ws;  ws += NF * sizeof(float);
    __bf16* wt = (__bf16*)ws; // 6 * 128*128 bf16

    __bf16* WtRel[3];
    __bf16* WtRoot[3];
    for (int l = 0; l < 3; ++l) {
        WtRel[l]  = wt + (size_t)(2 * l)     * DIM * DIM;
        WtRoot[l] = wt + (size_t)(2 * l + 1) * DIM * DIM;
    }

    // one-time (per launch) bf16 transpose of the 6 weight matrices
    for (int l = 0; l < 3; ++l) {
        convert_weight_kernel<<<64, 256, 0, stream>>>(W_rel[l],  WtRel[l]);
        convert_weight_kernel<<<64, 256, 0, stream>>>(W_root[l], WtRoot[l]);
    }

    const int* src = edge;
    const int* dst = edge + N_EDGES;

    const float* in = x;
    float* outs[3] = {h1, h2, h1};
    for (int l = 0; l < 3; ++l) {
        fill_kernel<<<4096, 256, 0, stream>>>(agg, 0.0f, (int)NF);
        scatter_add_kernel<<<(N_EDGES * 32) / 256, 256, 0, stream>>>(in, src, dst, agg);
        graphconv_gemm_kernel<<<N_NODES / 16, 256, 0, stream>>>(
            agg, in, WtRel[l], WtRoot[l], b_rel[l], outs[l], (l < 2) ? 1 : 0);
        in = outs[l];
    }

    fill_kernel<<<64, 256, 0, stream>>>((float*)d_out, -__builtin_inff(), N_GRAPHS * DIM);
    pool_max_kernel<<<(N_NODES * 32 + 255) / 256, 256, 0, stream>>>(in, batch, (float*)d_out);
}
